// MultiHeadAttention_46471546143554
// MI455X (gfx1250) — matrix operand order repaired and checked by reading, not compiled, not run on hardware
//
#include <hip/hip_runtime.h>
#include <cstdint>
#include <cstddef>

#define DEVI static __device__ __forceinline__

typedef __bf16 bf16;
typedef __attribute__((ext_vector_type(16))) __bf16 v16bf;
typedef __attribute__((ext_vector_type(8)))  float  v8f;

static constexpr int Bsz = 2, Tdim = 2048, Cdim = 1024, Hn = 16;

DEVI bf16 f2bf(float f) {
  union { float f; uint32_t u; } a; a.f = f;
  uint32_t r = (a.u + 0x7FFFu + ((a.u >> 16) & 1u)) >> 16;
  union { uint16_t s; bf16 b; } o; o.s = (uint16_t)r;
  return o.b;
}
DEVI float bf2f(bf16 b) {
  union { uint16_t s; bf16 b; } i; i.b = b;
  union { float f; uint32_t u; } o; o.u = ((uint32_t)i.s) << 16;
  return o.f;
}
DEVI v8f zero8() {
  v8f z;
#pragma unroll
  for (int i = 0; i < 8; ++i) z[i] = 0.f;
  return z;
}
// Assemble a 16-element bf16 fragment from two 16-byte runs in LDS.
DEVI v16bf ld16(const bf16* p0, const bf16* p1) {
  union { uint4 q[2]; v16bf v; } u;
  u.q[0] = *(const uint4*)p0;
  u.q[1] = *(const uint4*)p1;
  return u.v;
}
// CDNA5 async global->LDS copy (16B), tracked by ASYNCcnt, bypasses VGPRs.
DEVI void async_cp16(const bf16* lds_dst, const bf16* gsrc) {
  uint32_t l = (uint32_t)(uintptr_t)lds_dst;          // flat-LDS addr[31:0] == DS offset
  uint64_t g = (uint64_t)(uintptr_t)gsrc;
  asm volatile("global_load_async_to_lds_b128 %0, %1, off" :: "v"(l), "v"(g) : "memory");
}
// ASYNCcnt completes in order: wait<=N leaves the newest N copies in flight.
DEVI void wait_async0() { asm volatile("s_wait_asynccnt 0x0" ::: "memory"); }
DEVI void wait_async6() { asm volatile("s_wait_asynccnt 0x6" ::: "memory"); }
DEVI void wait_async8() { asm volatile("s_wait_asynccnt 0x8" ::: "memory"); }

// ---------------------------------------------------------------- converts
__global__ __launch_bounds__(256)
void f32_to_bf16_kernel(const float* __restrict__ in, bf16* __restrict__ out, int n) {
  int i = blockIdx.x * 256 + threadIdx.x;
  if (i < n) out[i] = f2bf(in[i]);
}
// out[n*K + k] = (bf16) in[k*N + n]  (weights pre-transposed to [N][K])
__global__ __launch_bounds__(256)
void f32_to_bf16_transpose_kernel(const float* __restrict__ in, bf16* __restrict__ out,
                                  int K, int N) {
  int g = blockIdx.x * 256 + threadIdx.x;
  if (g >= N * K) return;
  int n = g / K, k = g % K;
  out[g] = f2bf(in[(size_t)k * N + n]);
}

// ---------------------------------------------------------------- GEMM (bf16 WMMA, f32 acc)
// Block tile 128x256, 8 waves (2 in M x 4 in N), each wave 64x64, K-step 32.
// A is [M][K]; Bt is pre-transposed [N][K]: both tiles are straight async copies,
// double-buffered so global fetch overlaps WMMA.
template <bool OUT_F32>
__global__ __launch_bounds__(256)
void gemm_bf16_wmma(const bf16* __restrict__ A, const bf16* __restrict__ Bt,
                    void* __restrict__ Out, int M, int N, int K) {
  __shared__ __align__(16) bf16 Alds[2][128 * 32];   // [m][k]
  __shared__ __align__(16) bf16 Blds[2][256 * 32];   // [n][k]
  const int tid = threadIdx.x;
  const int wid = tid >> 5, lane = tid & 31;
  const int waveM = wid & 1, waveN = wid >> 1;
  const int hf = lane >> 4, l16 = lane & 15;
  const int rowBase = blockIdx.y * 128;
  const int colBase = blockIdx.x * 256;
  const int NI = K >> 5;

  auto issue = [&](int iter, int buf) {
    int k0 = iter << 5;
#pragma unroll
    for (int i = 0; i < 2; ++i) {          // A tile 128x32: 2 x b128 per thread
      int idx = tid + i * 256;
      int r = idx >> 2, c = idx & 3;
      async_cp16(&Alds[buf][r * 32 + c * 8], &A[(size_t)(rowBase + r) * K + k0 + c * 8]);
    }
#pragma unroll
    for (int i = 0; i < 4; ++i) {          // B tile 256x32: 4 x b128 per thread
      int idx = tid + i * 256;
      int r = idx >> 2, c = idx & 3;
      async_cp16(&Blds[buf][r * 32 + c * 8], &Bt[(size_t)(colBase + r) * K + k0 + c * 8]);
    }
  };

  v8f acc[4][4];
#pragma unroll
  for (int mt = 0; mt < 4; ++mt)
#pragma unroll
    for (int nt = 0; nt < 4; ++nt) acc[mt][nt] = zero8();

  issue(0, 0);
  for (int it = 0; it < NI; ++it) {
    const int p = it & 1;
    if (it + 1 < NI) { issue(it + 1, p ^ 1); wait_async6(); }   // prev 6 done, next 6 in flight
    else             { wait_async0(); }
    __syncthreads();

    v16bf af[4];
#pragma unroll
    for (int mt = 0; mt < 4; ++mt) {
      const bf16* ar = &Alds[p][(waveM * 64 + mt * 16 + l16) * 32];
      af[mt] = ld16(ar + hf * 8, ar + 16 + hf * 8);        // A layout: k runs split by lane half
    }
#pragma unroll
    for (int nt = 0; nt < 4; ++nt) {
      const bf16* br = &Blds[p][(waveN * 64 + nt * 16 + l16) * 32];
      v16bf bfr = ld16(br + hf * 8, br + 16 + hf * 8);     // B layout: k runs split by lane half
#pragma unroll
      for (int mt = 0; mt < 4; ++mt)
        acc[mt][nt] = __builtin_amdgcn_wmma_f32_16x16x32_bf16(
            false, af[mt], false, bfr, (short)0, acc[mt][nt], false, false);
    }
    __syncthreads();   // safe to overwrite buffer p next iteration
  }

  // D layout: row = r + 8*hf, col = l16
#pragma unroll
  for (int mt = 0; mt < 4; ++mt) {
#pragma unroll
    for (int nt = 0; nt < 4; ++nt) {
      int col = colBase + waveN * 64 + nt * 16 + l16;
      int row0 = rowBase + waveM * 64 + mt * 16 + hf * 8;
#pragma unroll
      for (int r = 0; r < 8; ++r) {
        size_t off = (size_t)(row0 + r) * N + col;
        if (OUT_F32) ((float*)Out)[off] = acc[mt][nt][r];
        else         ((bf16*)Out)[off] = f2bf(acc[mt][nt][r]);
      }
    }
  }
}

// ---------------------------------------------------------------- RoPE + head split
// Q,K -> [B,H,T,64]; V -> transposed [B,H,64,T] so attention V tiles are straight copies.
__global__ __launch_bounds__(256)
void rope_pack_kernel(const bf16* __restrict__ qkv,
                      const float* __restrict__ rsin, const float* __restrict__ rcos,
                      bf16* __restrict__ Qh, bf16* __restrict__ Kh, bf16* __restrict__ Vh) {
  int g = blockIdx.x * 256 + threadIdx.x;           // 2^21 threads total
  int d = g & 31;
  int h = (g >> 5) & 15;
  int t = (g >> 9) & 2047;
  int b = g >> 20;
  const bf16* row = qkv + (size_t)(b * Tdim + t) * (3 * Cdim);
  float q1 = bf2f(row[h * 64 + d]),            q2 = bf2f(row[h * 64 + d + 32]);
  float k1 = bf2f(row[Cdim + h * 64 + d]),     k2 = bf2f(row[Cdim + h * 64 + d + 32]);
  bf16  v1 = row[2 * Cdim + h * 64 + d],       v2 = row[2 * Cdim + h * 64 + d + 32];
  float s = rsin[t * 32 + d], c = rcos[t * 32 + d];
  size_t o = ((size_t)(b * Hn + h) * Tdim + t) * 64 + d;
  Qh[o] = f2bf(q1 * c - q2 * s);  Qh[o + 32] = f2bf(q1 * s + q2 * c);
  Kh[o] = f2bf(k1 * c - k2 * s);  Kh[o + 32] = f2bf(k1 * s + k2 * c);
  size_t ov = ((size_t)(b * Hn + h) * 64 + d) * Tdim + t;   // transposed V
  Vh[ov] = v1;
  Vh[ov + (size_t)32 * Tdim] = v2;
}

// ---------------------------------------------------------------- Flash attention (bf16 WMMA)
// Block = 4 waves; block handles 64 query rows of one (b,h); wave owns 16 rows.
// K/V chunks double-buffered through the async engine.
__global__ __launch_bounds__(128)
void flash_attn_wmma(const bf16* __restrict__ Qh, const bf16* __restrict__ Kh,
                     const bf16* __restrict__ Vh, bf16* __restrict__ Ob) {
  __shared__ __align__(16) bf16 Klds[2][64 * 64];   // [key][d]
  __shared__ __align__(16) bf16 Vt[2][64 * 64];     // [d][key]
  __shared__ __align__(16) bf16 Plds[4 * 16 * 64];  // per-wave P tile
  const int tid = threadIdx.x;
  const int wid = tid >> 5, lane = tid & 31;
  const int hf = lane >> 4, l16 = lane & 15;
  const int bid = blockIdx.x;
  const int qt = bid & 31;
  const int h  = (bid >> 5) & 15;
  const int b  = bid >> 9;
  const size_t headOff = (size_t)(b * Hn + h) * Tdim * 64;  // same offset for K and V-transposed

  auto issueKV = [&](int kc, int buf) {
    const bf16* kb = Kh + headOff + (size_t)(kc * 64) * 64;  // [key][d] chunk
    const bf16* vb = Vh + headOff + (size_t)(kc * 64);       // [d][T], col offset kc*64
#pragma unroll
    for (int i = 0; i < 4; ++i) {
      int idx = tid + i * 128;
      int r = idx >> 3, c8 = idx & 7;
      async_cp16(&Klds[buf][r * 64 + c8 * 8], &kb[r * 64 + c8 * 8]);
      async_cp16(&Vt[buf][r * 64 + c8 * 8], &vb[(size_t)r * Tdim + c8 * 8]);
    }
  };

  // Q fragments (A layout), resident for the whole kernel
  const bf16* qp = Qh + headOff + (size_t)(qt * 64 + wid * 16 + l16) * 64;
  v16bf qf[2];
#pragma unroll
  for (int s = 0; s < 2; ++s)
    qf[s] = ld16(qp + s * 32 + hf * 8, qp + s * 32 + 16 + hf * 8);

  v8f o[4];
#pragma unroll
  for (int dt = 0; dt < 4; ++dt) o[dt] = zero8();
  float mi[8], li[8];
#pragma unroll
  for (int r = 0; r < 8; ++r) { mi[r] = -3.0e38f; li[r] = 0.f; }

  issueKV(0, 0);
  for (int kc = 0; kc <= qt; ++kc) {
    const int p = kc & 1;
    if (kc + 1 <= qt) { issueKV(kc + 1, p ^ 1); wait_async8(); }  // prev 8 done, next 8 in flight
    else              { wait_async0(); }
    if (kc + 2 <= qt)                                             // L2 prefetch two chunks ahead
      __builtin_prefetch(Kh + headOff + (size_t)((kc + 2) * 64) * 64 + tid * 32, 0, 3);
    __syncthreads();

    // S = (Q K^T) * scale  (B fragment: n = key, k = d)
    v8f sa[4];
#pragma unroll
    for (int nt = 0; nt < 4; ++nt) {
      sa[nt] = zero8();
      const bf16* kr = &Klds[p][(nt * 16 + l16) * 64];
#pragma unroll
      for (int s = 0; s < 2; ++s) {
        v16bf kf = ld16(kr + s * 32 + hf * 8, kr + s * 32 + 16 + hf * 8);
        sa[nt] = __builtin_amdgcn_wmma_f32_16x16x32_bf16(
            false, qf[s], false, kf, (short)0, sa[nt], false, false);
      }
    }
#pragma unroll
    for (int nt = 0; nt < 4; ++nt)
#pragma unroll
      for (int r = 0; r < 8; ++r) sa[nt][r] *= 0.125f;   // hd^-0.5
    if (kc == qt) {                                      // diagonal chunk: causal mask
#pragma unroll
      for (int nt = 0; nt < 4; ++nt) {
        int key = nt * 16 + l16;
#pragma unroll
        for (int r = 0; r < 8; ++r)
          if (key > wid * 16 + r + 8 * hf) sa[nt][r] = -3.0e38f;
      }
    }

    // online softmax: row m = r + 8*hf lives in one 16-lane half; reduce over n via xor-shuffles
    float alpha[8];
#pragma unroll
    for (int r = 0; r < 8; ++r) {
      float rm = fmaxf(fmaxf(sa[0][r], sa[1][r]), fmaxf(sa[2][r], sa[3][r]));
#pragma unroll
      for (int m = 1; m <= 8; m <<= 1) rm = fmaxf(rm, __shfl_xor(rm, m, 32));
      float mn = fmaxf(mi[r], rm);
      alpha[r] = __expf(mi[r] - mn);
      mi[r] = mn;
      float rs = 0.f;
#pragma unroll
      for (int nt = 0; nt < 4; ++nt) { sa[nt][r] = __expf(sa[nt][r] - mn); rs += sa[nt][r]; }
#pragma unroll
      for (int m = 1; m <= 8; m <<= 1) rs += __shfl_xor(rs, m, 32);
      li[r] = li[r] * alpha[r] + rs;
    }
#pragma unroll
    for (int dt = 0; dt < 4; ++dt)
#pragma unroll
      for (int r = 0; r < 8; ++r) o[dt][r] *= alpha[r];

    // P: D-frag -> LDS -> A-frag (wave-local; LDS is in-order per wave)
    bf16* pw = &Plds[wid * 16 * 64];
#pragma unroll
    for (int nt = 0; nt < 4; ++nt)
#pragma unroll
      for (int r = 0; r < 8; ++r)
        pw[(r + 8 * hf) * 64 + nt * 16 + l16] = f2bf(sa[nt][r]);
    const bf16* prd = &pw[l16 * 64];
    v16bf pf[2];
#pragma unroll
    for (int s = 0; s < 2; ++s)
      pf[s] = ld16(prd + s * 32 + hf * 8, prd + s * 32 + 16 + hf * 8);

    // O += P V  (B fragment from transposed V tile: n = d, k = key)
#pragma unroll
    for (int dt = 0; dt < 4; ++dt) {
      const bf16* vr = &Vt[p][(dt * 16 + l16) * 64];
#pragma unroll
      for (int s = 0; s < 2; ++s) {
        v16bf vf = ld16(vr + s * 32 + hf * 8, vr + s * 32 + 16 + hf * 8);
        o[dt] = __builtin_amdgcn_wmma_f32_16x16x32_bf16(
            false, pf[s], false, vf, (short)0, o[dt], false, false);
      }
    }
    __syncthreads();   // buffer p may be overwritten next iteration
  }

  // write attn output [B*T, C] bf16 for the proj GEMM
#pragma unroll
  for (int dt = 0; dt < 4; ++dt) {
#pragma unroll
    for (int r = 0; r < 8; ++r) {
      int row = qt * 64 + wid * 16 + r + 8 * hf;
      int col = h * 64 + dt * 16 + l16;
      Ob[((size_t)b * Tdim + row) * Cdim + col] = f2bf(o[dt][r] / li[r]);
    }
  }
}

// ---------------------------------------------------------------- launch
extern "C" void kernel_launch(void* const* d_in, const int* in_sizes, int n_in,
                              void* d_out, int out_size, void* d_ws, size_t ws_size,
                              hipStream_t stream) {
  const float* x     = (const float*)d_in[0];
  const float* Wqkv  = (const float*)d_in[1];
  const float* Wproj = (const float*)d_in[2];
  const float* rsin  = (const float*)d_in[3];
  const float* rcos  = (const float*)d_in[4];

  char* ws = (char*)d_ws;
  size_t off = 0;
  bf16* xb     = (bf16*)(ws + off); off += (size_t)4096 * 1024 * 2;   // x bf16 [M][K]
  bf16* wqkvT  = (bf16*)(ws + off); off += (size_t)3072 * 1024 * 2;   // Wqkv^T bf16 [N][K]
  bf16* wprojT = (bf16*)(ws + off); off += (size_t)1024 * 1024 * 2;   // Wproj^T bf16 [N][K]
  bf16* qkvb   = (bf16*)(ws + off); off += (size_t)4096 * 3072 * 2;   // qkv bf16
  bf16* Qhb    = (bf16*)(ws + off); off += (size_t)2 * 16 * 2048 * 64 * 2;  // [B,H,T,64]
  bf16* Khb    = (bf16*)(ws + off); off += (size_t)2 * 16 * 2048 * 64 * 2;  // [B,H,T,64]
  bf16* Vhb    = (bf16*)(ws + off); off += (size_t)2 * 16 * 2048 * 64 * 2;  // [B,H,64,T]
  bf16* attnb  = (bf16*)(ws + off); off += (size_t)4096 * 1024 * 2;

  f32_to_bf16_kernel<<<4194304 / 256, 256, 0, stream>>>(x, xb, 4194304);
  f32_to_bf16_transpose_kernel<<<3145728 / 256, 256, 0, stream>>>(Wqkv, wqkvT, 1024, 3072);
  f32_to_bf16_transpose_kernel<<<1048576 / 256, 256, 0, stream>>>(Wproj, wprojT, 1024, 1024);

  gemm_bf16_wmma<false><<<dim3(3072 / 256, 4096 / 128), 256, 0, stream>>>(
      xb, wqkvT, (void*)qkvb, 4096, 3072, 1024);

  rope_pack_kernel<<<2097152 / 256, 256, 0, stream>>>(qkvb, rsin, rcos, Qhb, Khb, Vhb);

  flash_attn_wmma<<<2 * 16 * 32, 128, 0, stream>>>(Qhb, Khb, Vhb, attnb);

  gemm_bf16_wmma<true><<<dim3(1024 / 256, 4096 / 128), 256, 0, stream>>>(
      attnb, wprojT, d_out, 4096, 1024, 1024);
}
